// CustomFasterRCNNModel_61598420959613
// MI455X (gfx1250) — matrix operand order repaired and checked by reading, not compiled, not run on hardware
//
#include <hip/hip_runtime.h>
#include <hip/hip_bf16.h>

// ---------------------------------------------------------------------------
// q [256, 262144] f32, s [512, 262144] f32.
// Per-row kNN ordering == ordering of  s2[j] - 2*dot(q_i, s_j)  (sqrt + q2
// eliminated). GEMM (68.7 GFLOP) runs as bf16 WMMA with f32 accumulate to sit
// on the 23.3 TB/s HBM roof; split-K partials keep all WGPs busy (M*N tiny).
// Pipeline per K-stage: async DMA global->LDS (f32, ASYNCcnt) overlapped with
// the WMMAs of the previous stage; a one-pass LDS convert produces bf16 tiles
// so the WMMA stream is pure ds_load_b128 -> v_wmma (no VALU hazards).
// ---------------------------------------------------------------------------
#define NQ   256
#define NS   512
#define DD   262144           // 256*32*32
#define MAXK 10

#define BM 128
#define BN 128
#define BK 64                 // K per stage (two WMMA k-steps of 32)
#define SA 72                 // bf16 LDS row stride (144B -> conflict-free b128 frags)
#define FTILE (BM * BK)       // f32 elements per f32 tile (unpadded; fills are contiguous)
#define STILE (BM * SA)       // bf16 elements per bf16 tile

typedef __attribute__((ext_vector_type(16))) __bf16 v16bf;
typedef __attribute__((ext_vector_type(8)))  __bf16 v8bf;
typedef __attribute__((ext_vector_type(8)))  float  v8f;
typedef int v4i __attribute__((vector_size(4 * sizeof(int))));   // matches builtin param

#if defined(__has_builtin)
#  if __has_builtin(__builtin_amdgcn_global_load_async_to_lds_b128)
#    define HAVE_ASYNC_LDS 1
#  endif
#endif
#ifndef HAVE_ASYNC_LDS
#  define HAVE_ASYNC_LDS 0
#endif

#define AS1P(p) ((__attribute__((address_space(1))) v4i*)(p))
#define AS3P(p) ((__attribute__((address_space(3))) v4i*)(p))

// ---------------------------------------------------------------------------
// Kernel 1: s2[j] = sum_k s[j,k]^2
// ---------------------------------------------------------------------------
__global__ void s2_kernel(const float* __restrict__ s, float* __restrict__ s2) {
    const int row = blockIdx.x;
    const int tid = threadIdx.x;
    const float4* s4 = (const float4*)(s + (size_t)row * DD);
    float acc = 0.f;
    for (int i = tid; i < DD / 4; i += 256) {
        float4 v = s4[i];
        acc += v.x * v.x + v.y * v.y + v.z * v.z + v.w * v.w;
    }
    __shared__ float red[256];
    red[tid] = acc;
    __syncthreads();
    for (int off = 128; off > 0; off >>= 1) {
        if (tid < off) red[tid] += red[tid + off];
        __syncthreads();
    }
    if (tid == 0) s2[row] = red[0];
}

// ---------------------------------------------------------------------------
// Kernel 2: split-K bf16 WMMA GEMM partials.
// Block (x=N-tile, y=M-tile, z=K-chunk) -> 128x128 f32 partial of q.s^T.
// 512 threads = 16 waves (wave32), wave grid 4(M) x 4(N), 2x2 accum tiles/wave.
// Stage: [convert f32->bf16 in LDS] barrier [issue async fill of next f32
// tile] [WMMA from bf16 tiles] [wait async] barrier.
// ---------------------------------------------------------------------------
__global__ void __launch_bounds__(512)
gemm_partial_kernel(const float* __restrict__ q,
                    const float* __restrict__ s,
                    float* __restrict__ partials,
                    int Kc) {
    extern __shared__ __align__(16) char smemraw[];
    float*  FA = (float*)smemraw;             // [BM][BK] f32  (32 KB)
    float*  FB = FA + FTILE;                  // [BN][BK] f32  (32 KB)
    __bf16* SAt = (__bf16*)(FB + FTILE);      // [BM][SA] bf16 (18 KB)
    __bf16* SBt = SAt + STILE;                // [BN][SA] bf16 (18 KB)

    const int tid  = threadIdx.x;
    const int lane = tid & 31;           // wave32
    const int wave = tid >> 5;           // 0..15
    const int wm   = wave >> 2;          // 0..3 -> M offset wm*32
    const int wn   = wave & 3;           // 0..3 -> N offset wn*32
    const int l16  = lane & 15;
    const int g    = lane >> 4;          // lane group (K-split select)

    const int i0 = blockIdx.y * BM;
    const int j0 = blockIdx.x * BN;

    // Fill/convert geometry: 2048 float4 per 128x64 tile / 512 threads = 4.
    // Thread t owns rows r0 + 32*i (i=0..3), f32 columns [c4, c4+4).
    const int r0 = tid >> 4;             // 0..31
    const int c4 = (tid & 15) << 2;      // 0,4,...,60
    const uint32_t kb0 = (uint32_t)blockIdx.z * (uint32_t)Kc * 4u;
    uint32_t aoff = ((uint32_t)(i0 + r0) * DD + (uint32_t)c4) * 4u + kb0;  // bytes
    uint32_t boff = ((uint32_t)(j0 + r0) * DD + (uint32_t)c4) * 4u + kb0;  // bytes
    const char* qc = (const char*)q;
    const char* sc = (const char*)s;
    const int foff = r0 * BK + c4;       // f32 tile element offset (contiguous fills)

#if HAVE_ASYNC_LDS
    // Async fill: LDS DMA, zero staging registers, tracked with ASYNCcnt.
#define FILL()                                                                \
    do {                                                                      \
        _Pragma("unroll")                                                     \
        for (int i = 0; i < 4; ++i) {                                         \
            __builtin_amdgcn_global_load_async_to_lds_b128(                   \
                AS1P(qc + (aoff + (uint32_t)i * (DD * 128u))),                \
                AS3P(FA + foff + i * 32 * BK), 0, 0);                         \
            __builtin_amdgcn_global_load_async_to_lds_b128(                   \
                AS1P(sc + (boff + (uint32_t)i * (DD * 128u))),                \
                AS3P(FB + foff + i * 32 * BK), 0, 0);                         \
        }                                                                     \
        aoff += BK * 4u;                                                      \
        boff += BK * 4u;                                                      \
    } while (0)
#if __has_builtin(__builtin_amdgcn_s_wait_asynccnt)
#define WAIT_FILL() __builtin_amdgcn_s_wait_asynccnt(0)
#else
#define WAIT_FILL() asm volatile("s_wait_asynccnt 0" ::: "memory")
#endif
#else
    // Sync fallback: stage through 8 float4 (32 VGPRs), store f32 to LDS.
#define FILL()                                                                \
    do {                                                                      \
        float4 ta[4], tb[4];                                                  \
        _Pragma("unroll")                                                     \
        for (int i = 0; i < 4; ++i) {                                         \
            ta[i] = *(const float4*)(qc + (aoff + (uint32_t)i * (DD * 128u)));\
            tb[i] = *(const float4*)(sc + (boff + (uint32_t)i * (DD * 128u)));\
        }                                                                     \
        _Pragma("unroll")                                                     \
        for (int i = 0; i < 4; ++i) {                                         \
            *(float4*)&FA[foff + i * 32 * BK] = ta[i];                        \
            *(float4*)&FB[foff + i * 32 * BK] = tb[i];                        \
        }                                                                     \
        aoff += BK * 4u;                                                      \
        boff += BK * 4u;                                                      \
    } while (0)
#define WAIT_FILL() do {} while (0)
#endif

    v8f c[2][2];
    #pragma unroll
    for (int mt = 0; mt < 2; ++mt)
        #pragma unroll
        for (int nt = 0; nt < 2; ++nt)
            c[mt][nt] = (v8f){0.f, 0.f, 0.f, 0.f, 0.f, 0.f, 0.f, 0.f};

    // Prologue: DMA stage-0 f32 tiles into LDS.
    FILL();
    WAIT_FILL();
    __syncthreads();

    for (int kk = 0; kk < Kc; kk += BK) {
        // Phase 1: one-pass f32 -> bf16 convert in LDS (16 cvt_pk / thread).
        #pragma unroll
        for (int i = 0; i < 4; ++i) {
            const int row = r0 + 32 * i;
            float4 va = *(const float4*)&FA[row * BK + c4];
            float4 vb = *(const float4*)&FB[row * BK + c4];
            __bf16* da = &SAt[row * SA + c4];
            da[0] = (__bf16)va.x; da[1] = (__bf16)va.y;
            da[2] = (__bf16)va.z; da[3] = (__bf16)va.w;
            __bf16* db = &SBt[row * SA + c4];
            db[0] = (__bf16)vb.x; db[1] = (__bf16)vb.y;
            db[2] = (__bf16)vb.z; db[3] = (__bf16)vb.w;
        }
        __syncthreads();                 // bf16 tiles ready; f32 tiles now free

        const bool more = (kk + BK) < Kc;
        if (more) FILL();                // async refill flows under the WMMAs

        // Phase 2: pure ds_load_b128 -> v_wmma stream (no VALU hazards).
        #pragma unroll
        for (int k = 0; k < BK; k += 32) {
            // A frag: lane = row M (l16); K in [k+8g,+8) U [k+16+8g,+8)
            v16bf a[2];
            #pragma unroll
            for (int mt = 0; mt < 2; ++mt) {
                const __bf16* pa = SAt + (wm * 32 + mt * 16 + l16) * SA;
                v8bf lo = *(const v8bf*)(pa + k + 8 * g);
                v8bf hi = *(const v8bf*)(pa + k + 16 + 8 * g);
                a[mt] = __builtin_shufflevector(lo, hi, 0, 1, 2, 3, 4, 5, 6, 7,
                                                        8, 9, 10, 11, 12, 13, 14, 15);
            }
            // B frag: lane = col N (l16); K in [k+8g,+8) U [k+16+8g,+8)
            v16bf b[2];
            #pragma unroll
            for (int nt = 0; nt < 2; ++nt) {
                const __bf16* pb = SBt + (wn * 32 + nt * 16 + l16) * SA;
                v8bf lo = *(const v8bf*)(pb + k + 8 * g);
                v8bf hi = *(const v8bf*)(pb + k + 16 + 8 * g);
                b[nt] = __builtin_shufflevector(lo, hi, 0, 1, 2, 3, 4, 5, 6, 7,
                                                        8, 9, 10, 11, 12, 13, 14, 15);
            }
            #pragma unroll
            for (int mt = 0; mt < 2; ++mt)
                #pragma unroll
                for (int nt = 0; nt < 2; ++nt)
                    c[mt][nt] = __builtin_amdgcn_wmma_f32_16x16x32_bf16(
                        false, a[mt], false, b[nt], (short)0, c[mt][nt],
                        false, false);
        }

        if (more) WAIT_FILL();           // next f32 tiles landed
        __syncthreads();                 // also fences bf16 reads vs next convert
    }

    // Store partial C. C layout: VGPR r -> M = r + 8*(lane/16), N = lane%16.
    float* P = partials + (size_t)blockIdx.z * ((size_t)NQ * NS);
    #pragma unroll
    for (int mt = 0; mt < 2; ++mt)
        #pragma unroll
        for (int nt = 0; nt < 2; ++nt)
            #pragma unroll
            for (int r = 0; r < 8; ++r) {
                const int m = i0 + wm * 32 + mt * 16 + g * 8 + r;
                const int n = j0 + wn * 32 + nt * 16 + l16;
                P[(size_t)m * NS + n] = c[mt][nt][r];
            }
#undef FILL
#undef WAIT_FILL
}

// ---------------------------------------------------------------------------
// Kernel 3: score[i,j] = s2[j] - 2 * sum_chunks partial[c][i,j]
// ---------------------------------------------------------------------------
__global__ void reduce_kernel(const float* __restrict__ partials,
                              const float* __restrict__ s2,
                              float* __restrict__ score,
                              int nchunk) {
    const size_t idx = (size_t)blockIdx.x * 256 + threadIdx.x;   // < NQ*NS
    float acc = 0.f;
    for (int cch = 0; cch < nchunk; ++cch)
        acc += partials[(size_t)cch * (NQ * NS) + idx];
    score[idx] = s2[idx & (NS - 1)] - 2.f * acc;
}

// ---------------------------------------------------------------------------
// Kernel 4: per-row indices of 10 smallest scores (ties -> lower index,
// matching jax.lax.top_k on -dist).
// ---------------------------------------------------------------------------
__global__ void topk_kernel(const float* __restrict__ score, int* __restrict__ knn) {
    const int r = threadIdx.x;            // one thread per query row (256)
    const float* row = score + (size_t)r * NS;
    int sel[MAXK];
    for (int kk = 0; kk < MAXK; ++kk) {
        float best = 3.4e38f;
        int bj = 0;
        for (int j = 0; j < NS; ++j) {
            bool used = false;
            for (int t = 0; t < kk; ++t) used |= (sel[t] == j);
            if (used) continue;
            const float v = row[j];
            if (v < best) { best = v; bj = j; }   // strict < keeps smallest index on ties
        }
        sel[kk] = bj;
        knn[r * MAXK + kk] = bj;
    }
}

// ---------------------------------------------------------------------------
// Kernel 5: prefix -> losses -> best_k (first min) -> threshold -> qmask
// ---------------------------------------------------------------------------
__global__ void finalize_kernel(const int* __restrict__ knn, float* __restrict__ qmask) {
    __shared__ float cs[MAXK];
    __shared__ float thr_s;
    const int tid = threadIdx.x;          // 256 threads
    if (tid < MAXK) {
        float acc = 0.f;
        for (int r = 0; r < NQ; ++r) acc += (float)knn[r * MAXK + tid];
        cs[tid] = acc;
    }
    __syncthreads();
    if (tid == 0) {
        float pre = 0.f, bestL = 3.4e38f, pref[MAXK];
        int bestK = 1;
        for (int k = 0; k < MAXK; ++k) {
            pre += cs[k];
            pref[k] = pre;
            const float loss = pre / ((float)NQ * (float)(k + 1));
            if (loss < bestL) { bestL = loss; bestK = k + 1; }   // first minimum
        }
        thr_s = pref[bestK - 1] / ((float)NQ * (float)bestK);
    }
    __syncthreads();
    qmask[tid] = ((float)knn[tid * MAXK + 0] < thr_s) ? 1.f : 0.f;
}

// ---------------------------------------------------------------------------
// Kernel 6: out = q * qmask[row]   (float4; row uniform per block)
// ---------------------------------------------------------------------------
__global__ void mask_kernel(const float* __restrict__ q,
                            const float* __restrict__ qmask,
                            float* __restrict__ out) {
    const float m = qmask[blockIdx.x >> 8];                 // 256 blocks per row
    const size_t idx = (size_t)blockIdx.x * 256 + threadIdx.x;
    const float4 v = ((const float4*)q)[idx];
    float4 o;
    o.x = v.x * m; o.y = v.y * m; o.z = v.z * m; o.w = v.w * m;
    ((float4*)out)[idx] = o;
}

// ---------------------------------------------------------------------------
extern "C" void kernel_launch(void* const* d_in, const int* in_sizes, int n_in,
                              void* d_out, int out_size, void* d_ws, size_t ws_size,
                              hipStream_t stream) {
    const float* q = (const float*)d_in[0];   // [256, 262144]
    const float* s = (const float*)d_in[1];   // [512, 262144]
    float* out = (float*)d_out;               // [256, 262144]

    const size_t CM = (size_t)NQ * NS;        // 131072
    const size_t base_bytes = (CM + NS + NQ + 64) * sizeof(float) + (size_t)NQ * MAXK * sizeof(int);

    // Split-K factor: largest power of two <= 32 that fits in the workspace.
    int nchunk = 1;
    while (nchunk < 32 &&
           ((size_t)(nchunk * 2) * CM * sizeof(float) + base_bytes) <= ws_size)
        nchunk <<= 1;

    float* partials = (float*)d_ws;                       // [nchunk, NQ, NS]
    float* score    = partials + (size_t)nchunk * CM;     // [NQ, NS]
    float* s2       = score + CM;                         // [NS]
    float* qmask    = s2 + NS;                            // [NQ]
    int*   knn      = (int*)(qmask + NQ);                 // [NQ, MAXK]

    const int Kc = DD / nchunk;                           // multiple of BK for nchunk<=32

    s2_kernel<<<NS, 256, 0, stream>>>(s, s2);

    dim3 gg(NS / BN, NQ / BM, nchunk);
    const size_t shbytes = 2 * (size_t)FTILE * sizeof(float)      // f32 tiles (64 KB)
                         + 2 * (size_t)STILE * sizeof(__bf16);    // bf16 tiles (36 KB)
    gemm_partial_kernel<<<gg, 512, shbytes, stream>>>(q, s, partials, Kc);

    reduce_kernel<<<(unsigned)(CM / 256), 256, 0, stream>>>(partials, s2, score, nchunk);
    topk_kernel<<<1, NQ, 0, stream>>>(score, knn);
    finalize_kernel<<<1, NQ, 0, stream>>>(knn, qmask);
    mask_kernel<<<NQ * (DD / 4) / 256, 256, 0, stream>>>(q, qmask, out);
}
